// AG_EAGCN_17343077941482
// MI455X (gfx1250) — matrix operand order repaired and checked by reading, not compiled, not run on hardware
//
#include <hip/hip_runtime.h>

#define NB 8
#define CC 256
#define HW 1024
#define NS 64

typedef __bf16 bf16_t;
typedef __attribute__((ext_vector_type(16))) __bf16 v16bf;
typedef __attribute__((ext_vector_type(8)))  float  v8f;

// ---------------------------------------------------------------------------
// Generic bf16 WMMA GEMM: C[M,N] = epilogue( A[M,K] @ B[K,N] ), batched (grid.y)
// One 32x32 macro-tile per wave (2x2 of 16x16 WMMA tiles) -> each A/B fragment
// feeds two v_wmma ops, halving operand traffic vs one-tile-per-wave.
// A/B row-major bf16, M and N multiples of 32, K multiple of 32.
// ---------------------------------------------------------------------------
__global__ __launch_bounds__(256) void k_gemm_bf16(
    const bf16_t* __restrict__ A, long long sA, int lda,
    const bf16_t* __restrict__ B, long long sB, int ldb,
    int M, int N, int K,
    const float* __restrict__ bias,                       // per-row m, unbatched
    const float* __restrict__ rowScale, long long sRow,   // per m
    const float* __restrict__ colScale, long long sCol,   // per n
    const float* __restrict__ addMat, long long sAdd, int ldadd,
    float accScale, float addScale, int doRelu,
    float* __restrict__ outF, long long sOutF,
    bf16_t* __restrict__ outB, long long sOutB, int ldc)
{
  const int lane = threadIdx.x & 31;
  const int wave = threadIdx.x >> 5;
  const int tilesM = M >> 5, tilesN = N >> 5;      // 32x32 macro tiles
  const int tile = blockIdx.x * 8 + wave;
  if (tile >= tilesM * tilesN) return;
  const int tm = tile % tilesM;          // consecutive waves share tn -> B reuse
  const int tn = tile / tilesM;
  const int z = blockIdx.y;
  const int m0 = tm << 5, n0 = tn << 5;

  const bf16_t* Ab = A + (long long)z * sA;
  const bf16_t* Bb = B + (long long)z * sB;
  const int r = lane & 15, half = lane >> 4;

  v8f acc00 = {0.f,0.f,0.f,0.f,0.f,0.f,0.f,0.f};
  v8f acc01 = acc00, acc10 = acc00, acc11 = acc00;

  for (int k0 = 0; k0 < K; k0 += 32) {
    // A fragments, 16x32 bf16 each: lane holds row r; half selects K 0-7/16-23
    // vs 8-15/24-31 within the 32-wide K panel.
    v16bf a0, a1, b0, b1;
    const bf16_t* ap0 = Ab + (long long)(m0 + r) * lda + k0 + half * 8;
    const bf16_t* ap1 = ap0 + (long long)16 * lda;
#pragma unroll
    for (int j = 0; j < 8; ++j) {
      a0[j] = ap0[j]; a0[j + 8] = ap0[16 + j];
      a1[j] = ap1[j]; a1[j + 8] = ap1[16 + j];
    }
    // B fragments, 32x16 bf16 each: lane holds column r; half selects K 0-7/16-23
    // vs 8-15/24-31 within the 32-wide K panel (same order as A).
    const bf16_t* bp = Bb + (long long)(k0 + half * 8) * ldb + n0 + r;
#pragma unroll
    for (int j = 0; j < 8; ++j) {
      b0[j] = bp[(long long)j * ldb]; b0[j + 8] = bp[(long long)(16 + j) * ldb];
      b1[j] = bp[(long long)j * ldb + 16]; b1[j + 8] = bp[(long long)(16 + j) * ldb + 16];
    }
    if (k0 + 32 < K) __builtin_prefetch(bp + 32 * ldb, 0, 3);

    acc00 = __builtin_amdgcn_wmma_f32_16x16x32_bf16(false, a0, false, b0, (short)0, acc00, false, false);
    acc01 = __builtin_amdgcn_wmma_f32_16x16x32_bf16(false, a0, false, b1, (short)0, acc01, false, false);
    acc10 = __builtin_amdgcn_wmma_f32_16x16x32_bf16(false, a1, false, b0, (short)0, acc10, false, false);
    acc11 = __builtin_amdgcn_wmma_f32_16x16x32_bf16(false, a1, false, b1, (short)0, acc11, false, false);
  }

  // C/D layout per 16x16 tile: VGPR g -> (row = g + 8*half, col = lane&15)
  v8f* accs[2][2] = {{&acc00, &acc01}, {&acc10, &acc11}};
#pragma unroll
  for (int ni = 0; ni < 2; ++ni) {
    const int col = n0 + (ni << 4) + (lane & 15);
    const float cs = colScale ? colScale[(long long)z * sCol + col] : 1.f;
#pragma unroll
    for (int mi = 0; mi < 2; ++mi) {
      const v8f& acc = *accs[mi][ni];
#pragma unroll
      for (int g = 0; g < 8; ++g) {
        const int row = m0 + (mi << 4) + g + (half << 3);
        float y = acc[g] * accScale;
        if (bias)     y += bias[row];
        if (doRelu)   y = fmaxf(y, 0.f);
        if (rowScale) y *= rowScale[(long long)z * sRow + row];
        if (colScale) y *= cs;
        if (addMat)   y += addScale * addMat[(long long)z * sAdd + (long long)row * ldadd + col];
        const long long o = (long long)row * ldc + col;
        if (outF) outF[(long long)z * sOutF + o] = y;
        if (outB) outB[(long long)z * sOutB + o] = (bf16_t)y;
      }
    }
  }
}

// ---------------------------------------------------------------------------
// helpers
// ---------------------------------------------------------------------------
__device__ __forceinline__ float blockReduce(float v, float* sm) {
  const int t = threadIdx.x;
  sm[t] = v; __syncthreads();
  for (int s = 128; s > 0; s >>= 1) { if (t < s) sm[t] += sm[t + s]; __syncthreads(); }
  v = sm[0]; __syncthreads();
  return v;
}

__global__ void k_f2bf(const float* __restrict__ in, bf16_t* __restrict__ out, long long n) {
  long long i = (long long)blockIdx.x * 256 + threadIdx.x;
  if (i < n) out[i] = (bf16_t)in[i];
}

__global__ void k_zero(float* p, int n) {
  int i = blockIdx.x * blockDim.x + threadIdx.x;
  if (i < n) p[i] = 0.f;
}

// pooled[n*C] = mean over HW of seg_cur
__global__ __launch_bounds__(256) void k_pool(const float* __restrict__ seg, float* __restrict__ pooled) {
  __shared__ float sm[256];
  const float* p = seg + (long long)blockIdx.x * HW;
  float s = 0.f;
  for (int i = threadIdx.x; i < HW; i += 256) s += p[i];
  s = blockReduce(s, sm);
  if (threadIdx.x == 0) pooled[blockIdx.x] = s * (1.f / HW);
}

// ca[n*S] = relu(wmlp @ pooled + bmlp)
__global__ void k_ca(const float* __restrict__ pooled, const float* __restrict__ wmlp,
                     const float* __restrict__ bmlp, float* __restrict__ ca) {
  const int nn = blockIdx.x, s = threadIdx.x;
  const float* pv = pooled + nn * CC;
  const float* wv = wmlp + s * CC;
  float a = bmlp[s];
  for (int c = 0; c < CC; ++c) a += pv[c] * wv[c];
  ca[nn * NS + s] = fmaxf(a, 0.f);
}

// u,v (rank-1 channel-attention factors, raw-view semantics) and
// seg_ss = ws2*max_c(seg)+bs2, edge_mm = ws3*max_c(seg*edge)+bs3
__global__ __launch_bounds__(256) void k_uvmax(
    const float* __restrict__ segc, const float* __restrict__ edge,
    const bf16_t* __restrict__ segsB, const float* __restrict__ ca,
    const float* ws2p, const float* bs2p, const float* ws3p, const float* bs3p,
    float* __restrict__ u, float* __restrict__ v,
    float* __restrict__ ssv, float* __restrict__ emv)
{
  const long long gid = (long long)blockIdx.x * 256 + threadIdx.x;
  const int nn = (int)(gid / HW), j = (int)(gid % HW);
  const bf16_t* ss = segsB + (long long)nn * NS * HW;
  const float* cap = ca + nn * NS;
  float ua = 0.f, va = 0.f;
  for (int s = 0; s < NS; ++s) {
    ua += (float)ss[(long long)j * NS + s] * cap[s];   // theta raw-view [HW,S]
    va += cap[s] * (float)ss[(long long)s * HW + j];   // theta_T natural [S,HW]
  }
  const float* sc = segc + (long long)nn * CC * HW;
  const float* ed = edge + (long long)nn * CC * HW;
  float m1 = -3.4e38f, m2 = -3.4e38f;
  for (int c = 0; c < CC; ++c) {
    const float x = sc[(long long)c * HW + j];
    m1 = fmaxf(m1, x);
    m2 = fmaxf(m2, x * ed[(long long)c * HW + j]);
  }
  u[gid] = ua; v[gid] = va;
  ssv[gid] = ws2p[0] * m1 + bs2p[0];
  emv[gid] = ws3p[0] * m2 + bs3p[0];
}

__device__ __forceinline__ float thr15(float x) { return x < 0.15f ? 0.f : x; }

// rowsums of symmetrized+thresholded sim_c (rank-1, on the fly) and sim_s (rawS)
__global__ __launch_bounds__(256) void k_rowcolsum(
    const float* __restrict__ rawS, const float* __restrict__ u, const float* __restrict__ v,
    float* __restrict__ e1, float* __restrict__ e2)
{
  __shared__ float sm[256];
  const int nn = blockIdx.x / HW, p = blockIdx.x % HW;
  const float* S = rawS + (long long)nn * HW * HW;
  const float* un = u + nn * HW;
  const float* vn = v + nn * HW;
  const float up = un[p], vp = vn[p];
  float rc = 0.f, cc = 0.f, rs = 0.f, cs = 0.f;
  for (int q = threadIdx.x; q < HW; q += 256) {
    rc += thr15(up * vn[q]);
    cc += thr15(un[q] * vp);
    rs += thr15(S[(long long)p * HW + q]);
    cs += thr15(S[(long long)q * HW + p]);
  }
  rc = blockReduce(rc, sm); cc = blockReduce(cc, sm);
  rs = blockReduce(rs, sm); cs = blockReduce(cs, sm);
  if (threadIdx.x == 0) {
    const float d1 = 0.5f * (rc + cc) + 1.f;
    const float d2 = 0.5f * (rs + cs) + 1.f;
    e1[blockIdx.x] = d1 > 0.f ? rsqrtf(d1) : 0.f;
    e2[blockIdx.x] = d2 > 0.f ? rsqrtf(d2) : 0.f;
  }
}

// pre = norm(sim_c) + norm(sim_s) + I ; rowsum -> e3 (pre is symmetric)
__global__ __launch_bounds__(256) void k_combine(
    const float* __restrict__ rawS, const float* __restrict__ u, const float* __restrict__ v,
    const float* __restrict__ e1, const float* __restrict__ e2,
    float* __restrict__ simPre, float* __restrict__ e3)
{
  __shared__ float sm[256];
  const int nn = blockIdx.x / HW, p = blockIdx.x % HW;
  const float* S = rawS + (long long)nn * HW * HW;
  float* O = simPre + (long long)nn * HW * HW + (long long)p * HW;
  const float* un = u + nn * HW;  const float* vn = v + nn * HW;
  const float* e1n = e1 + nn * HW; const float* e2n = e2 + nn * HW;
  const float up = un[p], vp = vn[p], e1p = e1n[p], e2p = e2n[p];
  float acc = 0.f;
  for (int q = threadIdx.x; q < HW; q += 256) {
    const float dlt = (q == p) ? 1.f : 0.f;
    const float Bc = 0.5f * (thr15(up * vn[q]) + thr15(un[q] * vp)) + dlt;
    const float Bs = 0.5f * (thr15(S[(long long)p * HW + q]) +
                             thr15(S[(long long)q * HW + p])) + dlt;
    const float pre = Bc * e1p * e1n[q] + Bs * e2p * e2n[q] + dlt;
    O[q] = pre;
    acc += pre;
  }
  acc = blockReduce(acc, sm);
  if (threadIdx.x == 0) e3[blockIdx.x] = acc > 0.f ? rsqrtf(acc) : 0.f;
}

// sim = pre * e3_p * e3_q ; write bf16 for GEMM, accumulate f32 into adjsum
__global__ __launch_bounds__(256) void k_normsim(
    const float* __restrict__ simPre, const float* __restrict__ e3,
    bf16_t* __restrict__ simB, float* __restrict__ adjsum, int accum)
{
  const long long gid = (long long)blockIdx.x * 256 + threadIdx.x;
  const long long per = (long long)HW * HW;
  const int nn = (int)(gid / per);
  const long long r = gid % per;
  const int p = (int)(r / HW), q = (int)(r % HW);
  const float val = simPre[gid] * e3[nn * HW + p] * e3[nn * HW + q];
  simB[gid] = (bf16_t)val;
  adjsum[gid] = accum ? (adjsum[gid] + val) : val;
}

// rowsum of (adjsum/3 + I) -> e4
__global__ __launch_bounds__(256) void k_rowsum_final(const float* __restrict__ adjsum,
                                                      float* __restrict__ e4) {
  __shared__ float sm[256];
  const int nn = blockIdx.x / HW, p = blockIdx.x % HW;
  const float* row = adjsum + (long long)nn * HW * HW + (long long)p * HW;
  float s = 0.f;
  for (int q = threadIdx.x; q < HW; q += 256) s += row[q];
  s = blockReduce(s, sm);
  if (threadIdx.x == 0) {
    const float d = s * (1.f / 3.f) + 1.f;
    e4[blockIdx.x] = d > 0.f ? rsqrtf(d) : 0.f;
  }
}

// final adj (bf16 for APPNP) + regularizer reductions: acc[0]=trace, acc[1]=sum a^2,
// acc[2+n]=sum_p log(rowsum)
__global__ __launch_bounds__(256) void k_final_adj(
    const float* __restrict__ adjsum, const float* __restrict__ e4,
    bf16_t* __restrict__ adjB, float* __restrict__ acc)
{
  __shared__ float sm[256];
  const int nn = blockIdx.x / HW, p = blockIdx.x % HW;
  const float* row = adjsum + (long long)nn * HW * HW + (long long)p * HW;
  bf16_t* orow = adjB + (long long)nn * HW * HW + (long long)p * HW;
  const float* e4n = e4 + nn * HW;
  const float e4p = e4n[p];
  float rs = 0.f, sq = 0.f, tr = 0.f;
  for (int q = threadIdx.x; q < HW; q += 256) {
    const float a = (row[q] * (1.f / 3.f) + (q == p ? 1.f : 0.f)) * e4p * e4n[q];
    orow[q] = (bf16_t)a;
    rs += a; sq += a * a;
    if (q == p) tr = a;
  }
  rs = blockReduce(rs, sm);
  sq = blockReduce(sq, sm);
  tr = blockReduce(tr, sm);
  if (threadIdx.x == 0) {
    atomicAdd(&acc[2 + nn], logf(rs));
    atomicAdd(&acc[1], sq);
    atomicAdd(&acc[0], tr);
  }
}

// t = relu(relu(x@wa1.T+ba1)@wa2.T+ba2), x = original seg raw-viewed [n,HW,C]
__global__ __launch_bounds__(256) void k_teleport(
    const float* __restrict__ seg0,
    const float* __restrict__ wa1, const float* __restrict__ ba1,
    const float* __restrict__ wa2, const float* __restrict__ ba2,
    float* __restrict__ tF, bf16_t* __restrict__ curB)
{
  __shared__ float sm[256];
  __shared__ float h[3];
  const int nn = blockIdx.x / HW, p = blockIdx.x % HW;
  const float* x = seg0 + (long long)nn * CC * HW + (long long)p * CC;
  const int k = threadIdx.x;                 // CC == blockDim
  const float xv = x[k];
  for (int j = 0; j < 3; ++j) {
    const float a = blockReduce(xv * wa1[j * CC + k], sm);
    if (threadIdx.x == 0) h[j] = fmaxf(a + ba1[j], 0.f);
  }
  __syncthreads();
  float tv = h[0] * wa2[k * 3 + 0] + h[1] * wa2[k * 3 + 1] + h[2] * wa2[k * 3 + 2] + ba2[k];
  tv = fmaxf(tv, 0.f);
  const long long o = (long long)nn * HW * CC + (long long)p * CC + k;
  tF[o] = tv;
  curB[o] = (bf16_t)tv;
}

// reg[n] = 0.1*trace_total + (-0.1*sum_log[n]/N + 0.1*sqrt(sum_sq_total)/N^2)
__global__ void k_reg(const float* __restrict__ acc, float* __restrict__ out) {
  const int i = threadIdx.x;  // 0..7
  out[i] = 0.1f * acc[0]
         + (-0.1f * acc[2 + i] * (1.f / 1024.f)
            + 0.1f * sqrtf(acc[1]) * (1.f / (1024.f * 1024.f)));
}

// ---------------------------------------------------------------------------
static void gemm(hipStream_t st,
                 const bf16_t* A, long long sA, int lda,
                 const bf16_t* B, long long sB, int ldb,
                 int M, int N, int K, int batch,
                 const float* bias,
                 const float* rsc, long long srs, const float* csc, long long scs,
                 const float* add, long long sadd, int ldadd,
                 float accS, float addS, int relu,
                 float* oF, long long soF, bf16_t* oB, long long soB, int ldc)
{
  const int tiles = (M / 32) * (N / 32);
  dim3 g((tiles + 7) / 8, batch);
  k_gemm_bf16<<<g, 256, 0, st>>>(A, sA, lda, B, sB, ldb, M, N, K, bias,
                                 rsc, srs, csc, scs, add, sadd, ldadd,
                                 accS, addS, relu, oF, soF, oB, soB, ldc);
}

extern "C" void kernel_launch(void* const* d_in, const int* in_sizes, int n_in,
                              void* d_out, int out_size, void* d_ws, size_t ws_size,
                              hipStream_t stream) {
  (void)in_sizes; (void)n_in; (void)out_size; (void)ws_size;
  const float* seg0 = (const float*)d_in[0];
  const float* edge = (const float*)d_in[1];
  const float* ws1  = (const float*)d_in[2];
  const float* bs1  = (const float*)d_in[3];
  const float* ws11 = (const float*)d_in[4];
  const float* bs11 = (const float*)d_in[5];
  const float* ws2  = (const float*)d_in[6];
  const float* bs2  = (const float*)d_in[7];
  const float* ws3  = (const float*)d_in[8];
  const float* bs3  = (const float*)d_in[9];
  const float* wmlp = (const float*)d_in[10];
  const float* bmlp = (const float*)d_in[11];
  const float* wgcn = (const float*)d_in[12];
  const float* bgcn = (const float*)d_in[13];
  const float* wa1  = (const float*)d_in[14];
  const float* ba1  = (const float*)d_in[15];
  const float* wa2  = (const float*)d_in[16];
  const float* ba2  = (const float*)d_in[17];
  float* outp = (float*)d_out;

  char* wsb = (char*)d_ws;
  size_t off = 0;
  auto alloc = [&](size_t bytes) -> void* {
    void* p = wsb + off;
    off += (bytes + 255) & ~(size_t)255;
    return p;
  };

  const long long SEG = (long long)NB * CC * HW;        // 2,097,152
  const long long ADJ = (long long)NB * HW * HW;        // 8,388,608

  float*  seg_cur  = (float*)alloc(SEG * 4);
  bf16_t* seg_bf   = (bf16_t*)alloc(SEG * 2);
  bf16_t* w1b      = (bf16_t*)alloc((size_t)NS * CC * 2);
  bf16_t* w11b     = (bf16_t*)alloc((size_t)NS * CC * 2);
  bf16_t* wgb      = (bf16_t*)alloc((size_t)CC * CC * 2);
  float*  pooled   = (float*)alloc((size_t)NB * CC * 4);
  float*  cav      = (float*)alloc((size_t)NB * NS * 4);
  bf16_t* segsB    = (bf16_t*)alloc((size_t)NB * NS * HW * 2);
  bf16_t* segcB    = (bf16_t*)alloc((size_t)NB * NS * HW * 2);
  float*  u        = (float*)alloc((size_t)NB * HW * 4);
  float*  v        = (float*)alloc((size_t)NB * HW * 4);
  float*  ssv      = (float*)alloc((size_t)NB * HW * 4);
  float*  emv      = (float*)alloc((size_t)NB * HW * 4);
  float*  e1       = (float*)alloc((size_t)NB * HW * 4);
  float*  e2       = (float*)alloc((size_t)NB * HW * 4);
  float*  e3       = (float*)alloc((size_t)NB * HW * 4);
  float*  e4       = (float*)alloc((size_t)NB * HW * 4);
  float*  rawS     = (float*)alloc(ADJ * 4);
  float*  simPre   = (float*)alloc(ADJ * 4);
  bf16_t* simB     = (bf16_t*)alloc(ADJ * 2);
  float*  adjsum   = (float*)alloc(ADJ * 4);
  bf16_t* segsimB  = (bf16_t*)alloc(SEG * 2);
  float*  acc      = (float*)alloc(16 * 4);

  // phase-2 buffers alias dead phase-1 regions
  bf16_t* adjB = (bf16_t*)simB;                              // 16 MiB
  float*  tF   = (float*)simPre;                             //  8 MiB
  bf16_t* curB = (bf16_t*)((char*)simPre + SEG * 4);         //  4 MiB
  bf16_t* nxtB = curB + SEG;                                 //  4 MiB

  // ---- init ----
  hipMemcpyAsync(seg_cur, seg0, SEG * 4, hipMemcpyDeviceToDevice, stream);
  k_f2bf<<<(int)((SEG + 255) / 256), 256, 0, stream>>>(seg0, seg_bf, SEG);
  k_f2bf<<<(NS * CC + 255) / 256, 256, 0, stream>>>(ws1, w1b, NS * CC);
  k_f2bf<<<(NS * CC + 255) / 256, 256, 0, stream>>>(ws11, w11b, NS * CC);
  k_f2bf<<<(CC * CC + 255) / 256, 256, 0, stream>>>(wgcn, wgb, CC * CC);

  // ---- 3 EAGCN propagation steps ----
  for (int step = 0; step < 3; ++step) {
    k_pool<<<NB * CC, 256, 0, stream>>>(seg_cur, pooled);
    k_ca<<<NB, NS, 0, stream>>>(pooled, wmlp, bmlp, cav);

    // 1x1 convs: [64,256] @ [256,1024] -> bf16
    gemm(stream, w1b, 0, CC, seg_bf, SEG / NB, HW, NS, HW, CC, NB,
         bs1, 0, 0, 0, 0, 0, 0, 0, 1.f, 0.f, 0,
         0, 0, segsB, (long long)NS * HW, HW);
    gemm(stream, w11b, 0, CC, seg_bf, SEG / NB, HW, NS, HW, CC, NB,
         bs11, 0, 0, 0, 0, 0, 0, 0, 1.f, 0.f, 0,
         0, 0, segcB, (long long)NS * HW, HW);

    k_uvmax<<<NB * HW / 256, 256, 0, stream>>>(seg_cur, edge, segsB, cav,
                                               ws2, bs2, ws3, bs3, u, v, ssv, emv);

    // sigma_out scaled by edge_mm (row) / seg_ss (col): [1024,64]@[64,1024]
    gemm(stream, segcB, (long long)NS * HW, NS, segcB, (long long)NS * HW, HW,
         HW, HW, NS, NB, 0, emv, HW, ssv, HW, 0, 0, 0, 1.f, 0.f, 0,
         rawS, (long long)HW * HW, 0, 0, HW);

    k_rowcolsum<<<NB * HW, 256, 0, stream>>>(rawS, u, v, e1, e2);
    k_combine<<<NB * HW, 256, 0, stream>>>(rawS, u, v, e1, e2, simPre, e3);
    k_normsim<<<(int)(ADJ / 256), 256, 0, stream>>>(simPre, e3, simB, adjsum,
                                                    step > 0 ? 1 : 0);

    // seg_sim = seg @ sim : [256,1024] @ [1024,1024] -> bf16
    gemm(stream, seg_bf, SEG / NB, HW, simB, (long long)HW * HW, HW,
         CC, HW, HW, NB, 0, 0, 0, 0, 0, 0, 0, 0, 1.f, 0.f, 0,
         0, 0, segsimB, SEG / NB, HW);

    // GCN: relu(wgcn @ seg_sim + bgcn) + seg -> new seg (f32 + bf16)
    gemm(stream, wgb, 0, CC, segsimB, SEG / NB, HW,
         CC, HW, CC, NB, bgcn, 0, 0, 0, 0,
         seg_cur, SEG / NB, HW, 1.f, 1.f, 1,
         seg_cur, SEG / NB, seg_bf, SEG / NB, HW);
  }

  // ---- final adjacency + regularizer ----
  k_zero<<<1, 16, 0, stream>>>(acc, 10);
  k_rowsum_final<<<NB * HW, 256, 0, stream>>>(adjsum, e4);
  k_final_adj<<<NB * HW, 256, 0, stream>>>(adjsum, e4, adjB, acc);

  // ---- APPNP ----
  k_teleport<<<NB * HW, CC, 0, stream>>>(seg0, wa1, ba1, wa2, ba2, tF, curB);

  // out = 0.5*adj@out + 0.5*t, three iterations; [1024,1024]@[1024,256]
  gemm(stream, adjB, (long long)HW * HW, HW, curB, SEG / NB, CC,
       HW, CC, HW, NB, 0, 0, 0, 0, 0, tF, SEG / NB, CC, 0.5f, 0.5f, 0,
       0, 0, nxtB, SEG / NB, CC);
  gemm(stream, adjB, (long long)HW * HW, HW, nxtB, SEG / NB, CC,
       HW, CC, HW, NB, 0, 0, 0, 0, 0, tF, SEG / NB, CC, 0.5f, 0.5f, 0,
       0, 0, curB, SEG / NB, CC);
  gemm(stream, adjB, (long long)HW * HW, HW, curB, SEG / NB, CC,
       HW, CC, HW, NB, 0, 0, 0, 0, 0, tF, SEG / NB, CC, 0.5f, 0.5f, 0,
       outp, SEG / NB, 0, 0, CC);

  k_reg<<<1, NB, 0, stream>>>(acc, outp + SEG);
}
